// EagleFlexMLA_32332513804774
// MI455X (gfx1250) — compile-verified, matrix operand order checked by reading, not run on hardware
//
#include <hip/hip_runtime.h>
#include <cstdint>
#include <cstddef>

#define H_    32
#define KVH_  8
#define D_    128
#define LAT_  256
#define B_    2
#define S_    1024
#define T_    (B_ * S_)      // 2048 tokens
#define H2_   8192
#define HD_   (H_ * D_)      // 4096
#define KD_   (KVH_ * D_)    // 1024

typedef __attribute__((ext_vector_type(16))) __bf16 bf16x16;
typedef __attribute__((ext_vector_type(8)))  float  f32x8;

union FragBF { uint4 q[2]; bf16x16 v; };

__device__ __forceinline__ uint16_t f2bf(float f) {
  uint32_t u = __float_as_uint(f);
  u += 0x7FFFu + ((u >> 16) & 1u);   // round-to-nearest-even
  return (uint16_t)(u >> 16);
}

__device__ __forceinline__ f32x8 wmma_bf16(FragBF a, FragBF b, f32x8 c) {
  return __builtin_amdgcn_wmma_f32_16x16x32_bf16(
      false, a.v, false, b.v, (short)0, c, false, false);
}

// CDNA5 async DMA: global -> LDS, tracked by ASYNCcnt (no VGPR round-trip).
// Generic LDS pointer low 32 bits == LDS byte offset (flat aperture model).
__device__ __forceinline__ void async_b128(const uint16_t* g, uint16_t* l) {
  asm volatile("global_load_async_to_lds_b128 %0, %1, off"
               :: "v"((uint32_t)(uintptr_t)l),
                  "v"((unsigned long long)(uintptr_t)g)
               : "memory");
}

__device__ __forceinline__ void wait_async0() {
#if __has_builtin(__builtin_amdgcn_s_wait_asynccnt)
  __builtin_amdgcn_s_wait_asynccnt(0);
#else
  asm volatile("s_wait_asynccnt 0" ::: "memory");
#endif
}

// ---------------------------------------------------------------------------
// fp32 -> bf16 pack
// ---------------------------------------------------------------------------
__global__ void cvt_f32_to_bf16(const float* __restrict__ x,
                                uint16_t* __restrict__ y, long n) {
  long i = (long)blockIdx.x * blockDim.x + threadIdx.x;
  long stride = (long)gridDim.x * blockDim.x;
  for (; i < n; i += stride) y[i] = f2bf(x[i]);
}

// ---------------------------------------------------------------------------
// C[M,N] (f32) = A[M,K](bf16) @ W[N,K](bf16)^T      M,N % 128 == 0, K % 32 == 0
// 256 threads / 8 waves; block tile 128x128; wave tile 64x32 (4x2 WMMA).
// Double-buffered LDS, async global->LDS DMA overlapped with WMMA.
// ---------------------------------------------------------------------------
__global__ __launch_bounds__(256)
void gemm_bf16_nt(const uint16_t* __restrict__ A,
                  const uint16_t* __restrict__ W,
                  float* __restrict__ C,
                  int M, int N, int K) {
  constexpr int LDT = 40;                 // padded LDS row stride (elems)
  __shared__ uint16_t ldsA[2][128 * LDT];
  __shared__ uint16_t ldsW[2][128 * LDT];
  const int tid  = threadIdx.x;
  const int lane = tid & 31;
  const int wave = tid >> 5;
  const int wm   = wave & 1;              // 2 wave-rows of 64
  const int wn   = wave >> 1;             // 4 wave-cols of 32
  const int bm   = blockIdx.x * 128;
  const int bn   = blockIdx.y * 128;
  const int lrow = lane & 15;
  const int koff = (lane >> 4) << 3;      // 0 or 8 (fragment K phase)
  const int rsel = (lane >> 4) << 3;      // C fragment row phase

  const f32x8 zero8 = {0.f, 0.f, 0.f, 0.f, 0.f, 0.f, 0.f, 0.f};
  f32x8 acc[4][2];
#pragma unroll
  for (int i = 0; i < 4; ++i)
#pragma unroll
    for (int j = 0; j < 2; ++j) acc[i][j] = zero8;

  auto stage = [&](int buf, int k0) {     // 128x32 slabs of A and W
#pragma unroll
    for (int i = 0; i < 2; ++i) {
      int idx = tid + (i << 8);           // 0..511
      int r   = idx >> 2;                 // 0..127
      int seg = (idx & 3) << 3;           // 0,8,16,24 elems
      async_b128(A + (size_t)(bm + r) * K + k0 + seg, &ldsA[buf][r * LDT + seg]);
      async_b128(W + (size_t)(bn + r) * K + k0 + seg, &ldsW[buf][r * LDT + seg]);
    }
  };

  stage(0, 0);
  wait_async0();
  __syncthreads();

  int buf = 0;
  for (int k0 = 0; k0 < K; k0 += 32) {
    if (k0 + 32 < K) stage(buf ^ 1, k0 + 32);  // DMA next slab during compute

    FragBF af[4], bfr[2];
#pragma unroll
    for (int mt = 0; mt < 4; ++mt) {
      const uint16_t* p = &ldsA[buf][(wm * 64 + mt * 16 + lrow) * LDT + koff];
      af[mt].q[0] = *(const uint4*)p;
      af[mt].q[1] = *(const uint4*)(p + 16);
    }
#pragma unroll
    for (int nt = 0; nt < 2; ++nt) {
      const uint16_t* p = &ldsW[buf][(wn * 32 + nt * 16 + lrow) * LDT + koff];
      bfr[nt].q[0] = *(const uint4*)p;
      bfr[nt].q[1] = *(const uint4*)(p + 16);
    }
#pragma unroll
    for (int mt = 0; mt < 4; ++mt)
#pragma unroll
      for (int nt = 0; nt < 2; ++nt)
        acc[mt][nt] = wmma_bf16(af[mt], bfr[nt], acc[mt][nt]);

    wait_async0();      // next buffer's DMA complete
    __syncthreads();    // all waves done reading current buffer
    buf ^= 1;
  }

#pragma unroll
  for (int mt = 0; mt < 4; ++mt)
#pragma unroll
    for (int nt = 0; nt < 2; ++nt) {
      int col = bn + wn * 32 + nt * 16 + lrow;
#pragma unroll
      for (int i = 0; i < 8; ++i) {
        int row = bm + wm * 64 + mt * 16 + rsel + i;
        C[(size_t)row * N + col] = acc[mt][nt][i];
      }
    }
}

// ---------------------------------------------------------------------------
// LayerNorm over LAT_=256, fp32 in -> bf16 out. One wave per token.
// ---------------------------------------------------------------------------
__global__ __launch_bounds__(256)
void ln_to_bf16(const float* __restrict__ x, const float* __restrict__ gamma,
                const float* __restrict__ beta, uint16_t* __restrict__ y, int T) {
  const int wave = threadIdx.x >> 5, lane = threadIdx.x & 31;
  const int t = blockIdx.x * 8 + wave;
  if (t >= T) return;
  const float* xp = x + (size_t)t * LAT_;
  float v[8];
  float s = 0.f;
#pragma unroll
  for (int j = 0; j < 8; ++j) { v[j] = xp[lane + 32 * j]; s += v[j]; }
#pragma unroll
  for (int o = 16; o; o >>= 1) s += __shfl_xor(s, o, 32);
  const float mu = s * (1.0f / LAT_);
  float var = 0.f;
#pragma unroll
  for (int j = 0; j < 8; ++j) { float d = v[j] - mu; var += d * d; }
#pragma unroll
  for (int o = 16; o; o >>= 1) var += __shfl_xor(var, o, 32);
  const float r = rsqrtf(var * (1.0f / LAT_) + 1e-5f);
#pragma unroll
  for (int j = 0; j < 8; ++j) {
    int c = lane + 32 * j;
    y[(size_t)t * LAT_ + c] = f2bf((v[j] - mu) * r * gamma[c] + beta[c]);
  }
}

// ---------------------------------------------------------------------------
// RoPE (full head dim, rotate-half) fp32 -> bf16.  cols = heads*D_ of tensor.
// ---------------------------------------------------------------------------
__global__ void rope_to_bf16(const float* __restrict__ x,
                             const int* __restrict__ pos,
                             uint16_t* __restrict__ y, long n, int cols) {
  long i = (long)blockIdx.x * blockDim.x + threadIdx.x;
  long stride = (long)gridDim.x * blockDim.x;
  for (; i < n; i += stride) {
    int t  = (int)(i / cols);
    int c  = (int)(i % cols);
    int dd = c & (D_ - 1);
    int k  = dd & 63;
    float ang = (float)pos[t] * __expf(-(float)k * 0.14391157f); // ln(1e4)/64
    float sn, cs;
    __sincosf(ang, &sn, &cs);
    float xv = x[i];
    float other = (dd < 64) ? -x[i + 64] : x[i - 64];
    y[i] = f2bf(xv * cs + other * sn);
  }
}

// ---------------------------------------------------------------------------
// Flash-style causal GQA attention, bf16 WMMA, f32 accumulators.
// grid = (S/128, H, B); block = 256 (8 waves); each wave owns 16 q rows.
// KV staged in 64-row tiles: ldsK [kv][d] (async DMA), ldsVt [d][kv] (manual
// transpose). P reshaped C-layout -> A-layout through a per-wave LDS slice.
// ---------------------------------------------------------------------------
__global__ __launch_bounds__(256)
void mla_attention(const uint16_t* __restrict__ Q,   // [T, HD_]  (RoPE'd)
                   const uint16_t* __restrict__ Kc,  // [T, KD_]  (RoPE'd)
                   const uint16_t* __restrict__ Vc,  // [T, KD_]
                   uint16_t* __restrict__ O) {       // [T, HD_]
  constexpr int LKT = 136;   // ldsK row stride (128 d + pad)
  constexpr int LVT = 72;    // ldsVt / ldsP row stride (64 kv + pad)
  __shared__ uint16_t ldsK[64 * LKT];
  __shared__ uint16_t ldsVt[128 * LVT];
  __shared__ uint16_t ldsP[8 * 16 * LVT];

  const int qblk = blockIdx.x;
  const int head = blockIdx.y;
  const int b    = blockIdx.z;
  const int kvh  = head >> 2;          // GQA: 4 q heads per kv head
  const int tid  = threadIdx.x;
  const int lane = tid & 31;
  const int wave = tid >> 5;
  const int lrow = lane & 15;
  const int koff = (lane >> 4) << 3;
  const int rsel = (lane >> 4) << 3;
  const int qrow0 = qblk * 128 + wave * 16;

  FragBF qf[4];
  {
    const uint16_t* qp = Q + (size_t)(b * S_ + qrow0 + lrow) * HD_ + head * D_;
#pragma unroll
    for (int dc = 0; dc < 4; ++dc) {
      qf[dc].q[0] = *(const uint4*)(qp + dc * 32 + koff);
      qf[dc].q[1] = *(const uint4*)(qp + dc * 32 + 16 + koff);
    }
  }

  const f32x8 zero8 = {0.f, 0.f, 0.f, 0.f, 0.f, 0.f, 0.f, 0.f};
  f32x8 oacc[8];
#pragma unroll
  for (int dt = 0; dt < 8; ++dt) oacc[dt] = zero8;
  float m8[8], l8[8];
#pragma unroll
  for (int i = 0; i < 8; ++i) { m8[i] = -INFINITY; l8[i] = 0.f; }

  const float scale = 0.08838834764831843f;  // 1/sqrt(128)
  const int nkv = 2 * qblk + 2;              // 64-row kv tiles under causal mask

  for (int kvt = 0; kvt < nkv; ++kvt) {
    const size_t kb = (size_t)(b * S_ + kvt * 64) * KD_ + kvh * D_;
#pragma unroll
    for (int i = 0; i < 4; ++i) {            // K via async DMA, V transposed
      int idx = tid + (i << 8);              // 0..1023
      int r   = idx >> 4;                    // kv row 0..63
      int seg = (idx & 15) << 3;             // d 0..120
      async_b128(Kc + kb + (size_t)r * KD_ + seg, ldsK + r * LKT + seg);
      uint4 vq = *(const uint4*)(Vc + kb + (size_t)r * KD_ + seg);
      const uint16_t* vs = (const uint16_t*)&vq;
#pragma unroll
      for (int j = 0; j < 8; ++j) ldsVt[(seg + j) * LVT + r] = vs[j];
    }
    wait_async0();
    __syncthreads();

    // S = Q K^T : 16 q rows x 64 kv cols (4 tiles x 4 d-chunks)
    f32x8 sc[4];
#pragma unroll
    for (int nt = 0; nt < 4; ++nt) {
      f32x8 s = zero8;
      const uint16_t* kp = ldsK + (nt * 16 + lrow) * LKT;
#pragma unroll
      for (int dc = 0; dc < 4; ++dc) {
        FragBF kf;
        kf.q[0] = *(const uint4*)(kp + dc * 32 + koff);
        kf.q[1] = *(const uint4*)(kp + dc * 32 + 16 + koff);
        s = wmma_bf16(qf[dc], kf, s);
      }
      sc[nt] = s;
    }

    const bool needMask = (kvt * 64 + 63) > qrow0;
#pragma unroll
    for (int nt = 0; nt < 4; ++nt) {
      int kcol = kvt * 64 + nt * 16 + lrow;
#pragma unroll
      for (int i = 0; i < 8; ++i) {
        float v = sc[nt][i] * scale;
        if (needMask && kcol > (qrow0 + rsel + i)) v = -INFINITY;
        sc[nt][i] = v;
      }
    }

    // online softmax: per-lane 8 rows; reduce across 16-lane halves
    float corr[8];
#pragma unroll
    for (int i = 0; i < 8; ++i) {
      float mx = fmaxf(fmaxf(sc[0][i], sc[1][i]), fmaxf(sc[2][i], sc[3][i]));
#pragma unroll
      for (int o = 8; o; o >>= 1) mx = fmaxf(mx, __shfl_xor(mx, o, 32));
      float mn = fmaxf(m8[i], mx);
      corr[i] = __expf(m8[i] - mn);
      m8[i] = mn;
    }
#pragma unroll
    for (int nt = 0; nt < 4; ++nt)
#pragma unroll
      for (int i = 0; i < 8; ++i) sc[nt][i] = __expf(sc[nt][i] - m8[i]);
#pragma unroll
    for (int i = 0; i < 8; ++i) {
      float rs = sc[0][i] + sc[1][i] + sc[2][i] + sc[3][i];
#pragma unroll
      for (int o = 8; o; o >>= 1) rs += __shfl_xor(rs, o, 32);
      l8[i] = l8[i] * corr[i] + rs;
    }
#pragma unroll
    for (int dt = 0; dt < 8; ++dt)
#pragma unroll
      for (int i = 0; i < 8; ++i) oacc[dt][i] *= corr[i];

    // reshape P (C-layout) -> A-layout via per-wave LDS slice
    uint16_t* pw = ldsP + wave * 16 * LVT;
#pragma unroll
    for (int nt = 0; nt < 4; ++nt)
#pragma unroll
      for (int i = 0; i < 8; ++i)
        pw[(rsel + i) * LVT + nt * 16 + lrow] = f2bf(sc[nt][i]);

    // O += P V : 2 kv-chunks x 8 d-tiles
#pragma unroll
    for (int kc = 0; kc < 2; ++kc) {
      FragBF pf;
      const uint16_t* pp = pw + lrow * LVT + kc * 32;
      pf.q[0] = *(const uint4*)(pp + koff);
      pf.q[1] = *(const uint4*)(pp + 16 + koff);
#pragma unroll
      for (int dt = 0; dt < 8; ++dt) {
        FragBF vf;
        const uint16_t* vp = ldsVt + (dt * 16 + lrow) * LVT + kc * 32;
        vf.q[0] = *(const uint4*)(vp + koff);
        vf.q[1] = *(const uint4*)(vp + 16 + koff);
        oacc[dt] = wmma_bf16(pf, vf, oacc[dt]);
      }
    }
    __syncthreads();
  }

#pragma unroll
  for (int dt = 0; dt < 8; ++dt) {
    int col = head * D_ + dt * 16 + lrow;
#pragma unroll
    for (int i = 0; i < 8; ++i) {
      int qr = qrow0 + rsel + i;
      O[(size_t)(b * S_ + qr) * HD_ + col] = f2bf(oacc[dt][i] / l8[i]);
    }
  }
}

// ---------------------------------------------------------------------------
extern "C" void kernel_launch(void* const* d_in, const int* in_sizes, int n_in,
                              void* d_out, int out_size, void* d_ws, size_t ws_size,
                              hipStream_t stream) {
  const float* hidden = (const float*)d_in[0];
  const int*   pos    = (const int*)d_in[1];
  const float* Wq     = (const float*)d_in[2];
  const float* Wdkv   = (const float*)d_in[3];
  const float* gamma  = (const float*)d_in[4];
  const float* beta   = (const float*)d_in[5];
  const float* Wuk    = (const float*)d_in[6];
  const float* Wuv    = (const float*)d_in[7];
  const float* Wo     = (const float*)d_in[8];
  float* out = (float*)d_out;

  char* w = (char*)d_ws;
  auto take = [&](size_t bytes) {
    void* p = (void*)w;
    w += (bytes + 255) & ~(size_t)255;
    return p;
  };

  uint16_t* hbf   = (uint16_t*)take((size_t)T_ * H2_ * 2);
  uint16_t* wqb   = (uint16_t*)take((size_t)HD_ * H2_ * 2);
  uint16_t* wdkvb = (uint16_t*)take((size_t)LAT_ * H2_ * 2);
  uint16_t* wukb  = (uint16_t*)take((size_t)KD_ * LAT_ * 2);
  uint16_t* wuvb  = (uint16_t*)take((size_t)KD_ * LAT_ * 2);
  uint16_t* wob   = (uint16_t*)take((size_t)HD_ * HD_ * 2);
  float*    qf    = (float*)take((size_t)T_ * HD_ * 4);
  float*    ckvf  = (float*)take((size_t)T_ * LAT_ * 4);
  uint16_t* ckvb  = (uint16_t*)take((size_t)T_ * LAT_ * 2);
  float*    kf    = (float*)take((size_t)T_ * KD_ * 4);
  float*    vf    = (float*)take((size_t)T_ * KD_ * 4);
  uint16_t* qb    = (uint16_t*)take((size_t)T_ * HD_ * 2);
  uint16_t* kb    = (uint16_t*)take((size_t)T_ * KD_ * 2);
  uint16_t* vb    = (uint16_t*)take((size_t)T_ * KD_ * 2);
  uint16_t* ab    = (uint16_t*)take((size_t)T_ * HD_ * 2);

  auto cvt = [&](const float* x, uint16_t* y, long n) {
    long want = (n + 255) / 256;
    int blk = (int)(want > 4096 ? 4096 : want);
    cvt_f32_to_bf16<<<blk, 256, 0, stream>>>(x, y, n);
  };
  cvt(hidden, hbf, (long)T_ * H2_);
  cvt(Wq,     wqb, (long)HD_ * H2_);
  cvt(Wdkv, wdkvb, (long)LAT_ * H2_);
  cvt(Wuk,   wukb, (long)KD_ * LAT_);
  cvt(Wuv,   wuvb, (long)KD_ * LAT_);
  cvt(Wo,     wob, (long)HD_ * HD_);

  // Q = hidden @ Wq^T ; c_kv = LN(hidden @ Wdkv^T)
  gemm_bf16_nt<<<dim3(T_ / 128, HD_ / 128), 256, 0, stream>>>(hbf, wqb, qf, T_, HD_, H2_);
  gemm_bf16_nt<<<dim3(T_ / 128, LAT_ / 128), 256, 0, stream>>>(hbf, wdkvb, ckvf, T_, LAT_, H2_);
  ln_to_bf16<<<T_ / 8, 256, 0, stream>>>(ckvf, gamma, beta, ckvb, T_);

  // K/V decompress
  gemm_bf16_nt<<<dim3(T_ / 128, KD_ / 128), 256, 0, stream>>>(ckvb, wukb, kf, T_, KD_, LAT_);
  gemm_bf16_nt<<<dim3(T_ / 128, KD_ / 128), 256, 0, stream>>>(ckvb, wuvb, vf, T_, KD_, LAT_);

  // RoPE(Q), RoPE(K), pack V
  rope_to_bf16<<<4096, 256, 0, stream>>>(qf, pos, qb, (long)T_ * HD_, HD_);
  rope_to_bf16<<<2048, 256, 0, stream>>>(kf, pos, kb, (long)T_ * KD_, KD_);
  cvt(vf, vb, (long)T_ * KD_);

  // causal GQA attention, then output projection
  mla_attention<<<dim3(S_ / 128, H_, B_), 256, 0, stream>>>(qb, kb, vb, ab);
  gemm_bf16_nt<<<dim3(T_ / 128, HD_ / 128), 256, 0, stream>>>(ab, wob, out, T_, HD_, HD_);

  (void)in_sizes; (void)n_in; (void)out_size; (void)ws_size;
}
